// TransformerForQuestionAnsweringConditional_38620345925763
// MI455X (gfx1250) — matrix operand order repaired and checked by reading, not compiled, not run on hardware
//
#include <hip/hip_runtime.h>
#include <hip/hip_bf16.h>
#include <math.h>

#define BB   32
#define SS   384
#define DD   768
#define KK   10
#define SEP  102
#define MM   (BB * SS)          // 12288 rows of sequence_output
#define MT   32                 // M-tile rows per workgroup in k_end (B-panel reuse x2)

typedef __attribute__((ext_vector_type(16))) __bf16 v16bf;
typedef __attribute__((ext_vector_type(8)))  __bf16 v8bf;
typedef __attribute__((ext_vector_type(8)))  float  v8f;

#define WMMA_BF16(A, Bm, C) \
  __builtin_amdgcn_wmma_f32_16x16x32_bf16(false, (A), false, (Bm), (short)0, (C), false, false)

// gfx1250 hardware tanh (TRANS unit) when available
__device__ __forceinline__ float fast_tanh(float x) {
#if __has_builtin(__builtin_amdgcn_tanhf)
  return __builtin_amdgcn_tanhf(x);
#else
  return tanhf(x);
#endif
}

// split fp32 into bf16 hi + bf16 residual (3-product scheme recovers ~fp32 GEMM accuracy)
__device__ __forceinline__ void bf16_split(float x, __bf16 &h, __bf16 &l) {
  h = (__bf16)x;
  l = (__bf16)(x - (float)h);
}

// A-fragment (16x32 bf16, wave32): lane<16 carries M=lane, K={kk+0..15};
// lane>=16 carries M=lane-16, K={kk+16..31}.  Caller pre-offsets base by
// row*DD + (half?16:0); we load the two 16B halves and concatenate.
__device__ __forceinline__ v16bf load_afrag(const __bf16* __restrict__ base) {
  v8bf a0 = *(const v8bf*)(base);
  v8bf a1 = *(const v8bf*)(base + 8);
  return __builtin_shufflevector(a0, a1, 0,1,2,3,4,5,6,7,8,9,10,11,12,13,14,15);
}

// ---------------- conversion kernels ----------------

__global__ void k_convert_A(const float* __restrict__ A,
                            __bf16* __restrict__ Ah, __bf16* __restrict__ Al) {
  int i = (blockIdx.x * blockDim.x + threadIdx.x) * 8;
  v8bf h, l;
#pragma unroll
  for (int j = 0; j < 8; ++j) {
    __bf16 hh, ll;
    bf16_split(A[i + j], hh, ll);
    h[j] = hh; l[j] = ll;
  }
  *(v8bf*)(Ah + i) = h;
  *(v8bf*)(Al + i) = l;
}

// transpose W_E (2*D x D, row-major over K) into N-major bf16 hi/lo panels:
// Wt[z][n*D + k] = split(W_E[(z*D + k)*D + n])
__global__ void k_convert_Wt(const float* __restrict__ WE,
                             __bf16* __restrict__ Wth, __bf16* __restrict__ Wtl) {
  __shared__ float tile[32][33];
  int z = blockIdx.z;
  int n0 = blockIdx.x * 32, k0 = blockIdx.y * 32;
  int tx = threadIdx.x, ty = threadIdx.y;
  tile[ty][tx] = WE[(size_t)(z * DD + k0 + ty) * DD + (n0 + tx)];
  __syncthreads();
  float v = tile[tx][ty];
  __bf16 h, l;
  bf16_split(v, h, l);
  size_t o = (size_t)z * DD * DD + (size_t)(n0 + ty) * DD + (k0 + tx);
  Wth[o] = h;
  Wtl[o] = l;
}

// ---------------- start logits (bandwidth-bound matvec + mask) ----------------

__global__ void k_start_logits(const float* __restrict__ seq,
                               const int* __restrict__ ids,
                               const int* __restrict__ tt,
                               const float* __restrict__ Wqa,
                               const float* __restrict__ bqa,
                               float* __restrict__ out) {
  int wave = threadIdx.x >> 5, lane = threadIdx.x & 31;
  int row = blockIdx.x * 8 + wave;
  const float* p = seq + (size_t)row * DD;
  float acc = 0.f;
  for (int j = lane; j < DD; j += 32) acc += p[j] * Wqa[2 * j];
#pragma unroll
  for (int m = 16; m >= 1; m >>= 1) acc += __shfl_xor(acc, m, 32);
  if (lane == 0) {
    bool mask = (tt[row] != 0) || (ids[row] == SEP);
    out[row] = mask ? -INFINITY : (acc + bqa[0]);
  }
}

// ---------------- per-batch top-k (iterative argmax, lower index wins ties) ----------------

__global__ void k_topk(const float* __restrict__ slog,
                       int* __restrict__ topk_ws,
                       float* __restrict__ out_tail) {
  __shared__ float mv[512];
  __shared__ float sv[512];
  __shared__ int   si[512];
  int b = blockIdx.x, t = threadIdx.x;
  mv[t] = (t < SS) ? slog[b * SS + t] : -INFINITY;
  __syncthreads();
  for (int k = 0; k < KK; ++k) {
    sv[t] = mv[t]; si[t] = t;
    __syncthreads();
    for (int off = 256; off >= 1; off >>= 1) {
      if (t < off) {
        float v1 = sv[t], v2 = sv[t + off];
        int   i1 = si[t], i2 = si[t + off];
        if (v2 > v1 || (v2 == v1 && i2 < i1)) { sv[t] = v2; si[t] = i2; }
      }
      __syncthreads();
    }
    if (t == 0) {
      int wi = si[0];
      topk_ws[b * KK + k] = wi;
      ((int*)out_tail)[b * KK + k] = wi;   // topk_idx output (int bits)
      mv[wi] = -INFINITY;
    }
    __syncthreads();
  }
}

// ---------------- rep_proj' = gather(seq, topk) @ W2 + b_E  (WMMA, split-bf16) ----------------

__global__ void __launch_bounds__(384)
k_repproj(const __bf16* __restrict__ Ah, const __bf16* __restrict__ Al,
          const __bf16* __restrict__ Wth, const __bf16* __restrict__ Wtl,
          const int* __restrict__ topk, const float* __restrict__ bE,
          float* __restrict__ rp) {
  int wave = threadIdx.x >> 5, lane = threadIdx.x & 31;
  int rloc = lane & 15, half = lane >> 4;
  int mt = blockIdx.x;                         // 20 tiles of 16 gathered rows (320 total)
  int r = mt * 16 + rloc;
  int b = r / KK, kq = r % KK;
  int arow = b * SS + topk[b * KK + kq];
  const __bf16* aph = Ah + (size_t)arow * DD + half * 16;
  const __bf16* apl = Al + (size_t)arow * DD + half * 16;
  const __bf16* W2h = Wth + (size_t)DD * DD;   // z==1 panel
  const __bf16* W2l = Wtl + (size_t)DD * DD;

  v8f c[4] = {};
  for (int kk = 0; kk < DD; kk += 32) {
    v16bf a_hi = load_afrag(aph + kk);
    v16bf a_lo = load_afrag(apl + kk);
#pragma unroll
    for (int t = 0; t < 4; ++t) {
      int ncol = (wave + t * 12) * 16 + rloc;
      const __bf16* bh = W2h + (size_t)ncol * DD + kk + half * 16;
      const __bf16* bl = W2l + (size_t)ncol * DD + kk + half * 16;
      v16bf b_hi = *(const v16bf*)bh;
      v16bf b_lo = *(const v16bf*)bl;
      c[t] = WMMA_BF16(a_hi, b_hi, c[t]);
      c[t] = WMMA_BF16(a_hi, b_lo, c[t]);
      c[t] = WMMA_BF16(a_lo, b_hi, c[t]);
    }
  }
#pragma unroll
  for (int t = 0; t < 4; ++t) {
    int n = (wave + t * 12) * 16 + rloc;
    float be = bE[n];
#pragma unroll
    for (int i = 0; i < 8; ++i) {
      int m = mt * 16 + i + half * 8;          // C layout: VGPR i -> M = i (+8 for hi half-wave)
      rp[(size_t)m * DD + n] = c[t][i] + be;
    }
  }
}

// ---------------- fused seq_proj GEMM (32-row tile) + tanh-dot epilogue -> end_logits ----------------

__global__ void __launch_bounds__(384)
k_end(const __bf16* __restrict__ Ah, const __bf16* __restrict__ Al,
      const __bf16* __restrict__ Wth, const __bf16* __restrict__ Wtl,
      const float* __restrict__ rp, const float* __restrict__ WEE,
      const float* __restrict__ bEE,
      const int* __restrict__ ids, const int* __restrict__ tt,
      float* __restrict__ out_end) {
  __shared__ float lds_seq[MT * DD];           // 96 KB seq_proj tile (CDNA5: 320KB LDS/WGP)
  __shared__ float lds_wee[DD];
  int tid = threadIdx.x;
  int wave = tid >> 5, lane = tid & 31;
  int rloc = lane & 15, half = lane >> 4;
  int m0 = blockIdx.x * MT;                    // 384 tiles of 32 rows; all rows share batch b

  for (int j = tid; j < DD; j += 384) lds_wee[j] = WEE[j];

  const __bf16* aph0 = Ah + (size_t)(m0 + rloc) * DD + half * 16;
  const __bf16* apl0 = Al + (size_t)(m0 + rloc) * DD + half * 16;
  const __bf16* aph1 = aph0 + (size_t)16 * DD;
  const __bf16* apl1 = apl0 + (size_t)16 * DD;

  v8f c[2][4] = {};
  for (int kk = 0; kk < DD; kk += 32) {
    __builtin_prefetch((const void*)(aph0 + kk + 64), 0, 0);  // speculative stream-ahead on A
    v16bf a_hi0 = load_afrag(aph0 + kk);
    v16bf a_lo0 = load_afrag(apl0 + kk);
    v16bf a_hi1 = load_afrag(aph1 + kk);
    v16bf a_lo1 = load_afrag(apl1 + kk);
#pragma unroll
    for (int t = 0; t < 4; ++t) {
      int ncol = (wave + t * 12) * 16 + rloc;
      const __bf16* bh = Wth + (size_t)ncol * DD + kk + half * 16;
      const __bf16* bl = Wtl + (size_t)ncol * DD + kk + half * 16;
      v16bf b_hi = *(const v16bf*)bh;
      v16bf b_lo = *(const v16bf*)bl;
      c[0][t] = WMMA_BF16(a_hi0, b_hi, c[0][t]);
      c[0][t] = WMMA_BF16(a_hi0, b_lo, c[0][t]);
      c[0][t] = WMMA_BF16(a_lo0, b_hi, c[0][t]);
      c[1][t] = WMMA_BF16(a_hi1, b_hi, c[1][t]);
      c[1][t] = WMMA_BF16(a_hi1, b_lo, c[1][t]);
      c[1][t] = WMMA_BF16(a_lo1, b_hi, c[1][t]);
    }
  }
#pragma unroll
  for (int sub = 0; sub < 2; ++sub)
#pragma unroll
    for (int t = 0; t < 4; ++t) {
      int n = (wave + t * 12) * 16 + rloc;
#pragma unroll
      for (int i = 0; i < 8; ++i)
        lds_seq[(sub * 16 + i + half * 8) * DD + n] = c[sub][t][i];
    }
  __syncthreads();

  int bB = m0 / SS;
  int s0 = m0 % SS;
  for (int o = wave; o < MT * KK; o += 12) {   // 320 (row, k) outputs per block
    int r = o / KK, kq = o % KK;
    const float* rpp = rp + (size_t)(bB * KK + kq) * DD;   // includes b_E already
    const float* sp  = lds_seq + r * DD;
    float acc = 0.f;
    for (int j = lane; j < DD; j += 32)
      acc += fast_tanh(sp[j] + rpp[j]) * lds_wee[j];
#pragma unroll
    for (int m = 16; m >= 1; m >>= 1) acc += __shfl_xor(acc, m, 32);
    if (lane == 0) {
      int s = s0 + r;
      int flat = bB * SS + s;
      bool mask = (tt[flat] != 0) || (ids[flat] == SEP);
      float v = acc + bEE[0];
      out_end[(size_t)(bB * KK + kq) * SS + s] = mask ? -INFINITY : v;
    }
  }
}

// ---------------- launcher ----------------

extern "C" void kernel_launch(void* const* d_in, const int* in_sizes, int n_in,
                              void* d_out, int out_size, void* d_ws, size_t ws_size,
                              hipStream_t stream) {
  (void)in_sizes; (void)n_in; (void)out_size; (void)ws_size;
  const float* seq = (const float*)d_in[0];
  const int*   ids = (const int*)  d_in[1];
  const int*   tt  = (const int*)  d_in[2];
  const float* Wqa = (const float*)d_in[3];
  const float* bqa = (const float*)d_in[4];
  const float* WE  = (const float*)d_in[5];
  const float* bE  = (const float*)d_in[6];
  const float* WEE = (const float*)d_in[7];
  const float* bEE = (const float*)d_in[8];
  float* out = (float*)d_out;

  char* ws = (char*)d_ws;
  __bf16* Ah  = (__bf16*)(void*)(ws);                 // 12288*768 bf16  (18.87 MB)
  __bf16* Al  = (__bf16*)(void*)(ws + 18874368);      // 18.87 MB
  __bf16* Wth = (__bf16*)(void*)(ws + 37748736);      // 2*768*768 bf16 (2.36 MB)
  __bf16* Wtl = (__bf16*)(void*)(ws + 40108032);      // 2.36 MB
  float*  rpj = (float*) (void*)(ws + 42467328);      // 320*768 f32 (0.98 MB)
  int*    tk  = (int*)   (void*)(ws + 43450368);      // 320 ints

  k_convert_A<<<(MM * DD) / (8 * 256), 256, 0, stream>>>(seq, Ah, Al);
  {
    dim3 tb(32, 32), tg(DD / 32, DD / 32, 2);
    k_convert_Wt<<<tg, tb, 0, stream>>>(WE, Wth, Wtl);
  }
  k_start_logits<<<MM / 8, 256, 0, stream>>>(seq, ids, tt, Wqa, bqa, out);
  k_topk<<<BB, 512, 0, stream>>>(out, tk, out + MM + BB * KK * SS);
  k_repproj<<<(BB * KK) / 16, 384, 0, stream>>>(Ah, Al, Wth, Wtl, tk, bE, rpj);
  k_end<<<MM / MT, 384, 0, stream>>>(Ah, Al, Wth, Wtl, rpj, WEE, bEE, ids, tt, out + MM);
}
